// AttentionModule_52578989638112
// MI455X (gfx1250) — matrix operand order repaired and checked by reading, not compiled, not run on hardware
//
#include <hip/hip_runtime.h>
#include <hip/hip_bf16.h>
#include <stdint.h>

typedef __bf16 bf16;
typedef __attribute__((ext_vector_type(2)))  __bf16 v2bf;
typedef __attribute__((ext_vector_type(16))) __bf16 v16bf;
typedef __attribute__((ext_vector_type(8)))  float  v8f;

// 16-byte vector type + address-space-qualified flavors for the async builtins.
// (Per hipcc diagnostic: param is 'vector_size(16) int __device__ *', global first.)
typedef int v4i_ __attribute__((vector_size(16)));
typedef __attribute__((address_space(1))) v4i_ gv4i_;   // global
typedef __attribute__((address_space(3))) v4i_ lv4i_;   // LDS

#define B_ 16
#define S_ 1024
#define E_ 512
#define D_ 512
#define H_ 8

static constexpr float kInvScale = 1.0f / 22.0f;   // SCALE = float(int(sqrt(512))) = 22
static constexpr float kEps = 1e-5f;

// Use gfx1250 async global->LDS copies (ASYNCcnt) when the toolchain exposes them.
#if defined(__has_builtin)
#  if __has_builtin(__builtin_amdgcn_global_load_async_to_lds_b128) && \
      __has_builtin(__builtin_amdgcn_s_wait_asynccnt)
#    define USE_ASYNC_LDS 1
#  endif
#endif
#ifndef USE_ASYNC_LDS
#  define USE_ASYNC_LDS 0
#endif

// ---------------------------------------------------------------------------
// WMMA fragment gathers from LDS (wave32 layouts per CDNA5 ISA 7.12.2)
// A (16x32, M on lanes): lanes 0-15 hold K 0-7 & 16-23, lanes 16-31 hold K 8-15 & 24-31
// B (32x16, N on lanes): lanes 0-15 hold K 0-7 & 16-23, lanes 16-31 hold K 8-15 & 24-31
// LDS rows use stride 40 elements (80 B) so 16-B accesses stay aligned.
// ---------------------------------------------------------------------------
__device__ __forceinline__ v16bf frag_a_lds(const bf16* base, int lane) {
  const int r  = lane & 15;
  const int kb = (lane >> 4) << 3;
  const bf16* p = base + r * 40 + kb;
  v16bf f;
#pragma unroll
  for (int pe = 0; pe < 8; ++pe) {
    const int k = (pe < 4) ? (pe * 2) : (16 + (pe - 4) * 2);
    v2bf pr = *(const v2bf*)(p + k);
    f[pe * 2]     = pr[0];
    f[pe * 2 + 1] = pr[1];
  }
  return f;
}

__device__ __forceinline__ v16bf frag_b_lds(const bf16* base, int lane) {
  const int n  = lane & 15;
  const int kb = (lane >> 4) << 3;
  const bf16* p = base + n * 40 + kb;
  v16bf f;
#pragma unroll
  for (int e = 0; e < 16; ++e) f[e] = p[(e < 8) ? e : (e + 8)];
  return f;
}

// ---------------------------------------------------------------------------
// 128x128 GEMM block tile, TRANSB layout only: operand (k,n) = Bg[n*ldb + k].
// 8 waves, each owns a 32x64 subtile (2x4 WMMA per 32-deep K-step).
// Double-buffered LDS, one barrier per K-step; stage of the next K-tile is
// issued before the compute so global latency overlaps the WMMAs
// (async global->LDS when available, register-staged otherwise).
// ---------------------------------------------------------------------------
template <class Epi>
__device__ __forceinline__ void gemm128x128(const bf16* __restrict__ Ag, int lda,
                                            const bf16* __restrict__ Bg, int ldb,
                                            int Kdim, Epi epi) {
  __shared__ __align__(16) bf16 As[2][128][40];
  __shared__ __align__(16) bf16 Bs[2][128][40];

  const int tid  = threadIdx.x;       // 0..255
  const int lane = tid & 31;
  const int wave = tid >> 5;
  const int wm = (wave >> 1) * 32;    // 4 waves along M
  const int wn = (wave & 1) * 64;     // 2 waves along N
  const int r  = tid >> 1;            // staging row 0..127
  const int c  = (tid & 1) * 16;      // staging col 0 or 16

  const bf16* aP = Ag + (long)r * lda + c;
  const bf16* bP = Bg + (long)r * ldb + c;

  v8f acc[2][4] = {};

  auto compute = [&](int buf) {
    v16bf af[2], bfv[4];
#pragma unroll
    for (int i = 0; i < 2; ++i) af[i]  = frag_a_lds(&As[buf][wm + 16 * i][0], lane);
#pragma unroll
    for (int j = 0; j < 4; ++j) bfv[j] = frag_b_lds(&Bs[buf][wn + 16 * j][0], lane);
#pragma unroll
    for (int i = 0; i < 2; ++i)
#pragma unroll
      for (int j = 0; j < 4; ++j)
        acc[i][j] = __builtin_amdgcn_wmma_f32_16x16x32_bf16(
            false, af[i], false, bfv[j], (short)0, acc[i][j], false, false);
  };

#if USE_ASYNC_LDS
  auto stage = [&](int buf, int k0) {
    __builtin_amdgcn_global_load_async_to_lds_b128(
        (gv4i_*)(aP + k0),     (lv4i_*)&As[buf][r][c],     0, 0);
    __builtin_amdgcn_global_load_async_to_lds_b128(
        (gv4i_*)(aP + k0 + 8), (lv4i_*)&As[buf][r][c + 8], 0, 0);
    __builtin_amdgcn_global_load_async_to_lds_b128(
        (gv4i_*)(bP + k0),     (lv4i_*)&Bs[buf][r][c],     0, 0);
    __builtin_amdgcn_global_load_async_to_lds_b128(
        (gv4i_*)(bP + k0 + 8), (lv4i_*)&Bs[buf][r][c + 8], 0, 0);
  };
  stage(0, 0);
  __builtin_amdgcn_s_wait_asynccnt(0);
  __syncthreads();
  int cur = 0;
  for (int k0 = 0; k0 < Kdim; k0 += 32) {
    const int nxt = cur ^ 1;
    if (k0 + 32 < Kdim) stage(nxt, k0 + 32);   // overlaps with compute below
    compute(cur);
    __builtin_amdgcn_s_wait_asynccnt(0);
    __syncthreads();
    cur = nxt;
  }
#else
  {
    uint4 a0 = *(const uint4*)(aP),     a1 = *(const uint4*)(aP + 8);
    uint4 b0 = *(const uint4*)(bP),     b1 = *(const uint4*)(bP + 8);
    *(uint4*)&As[0][r][c] = a0;  *(uint4*)&As[0][r][c + 8] = a1;
    *(uint4*)&Bs[0][r][c] = b0;  *(uint4*)&Bs[0][r][c + 8] = b1;
  }
  __syncthreads();
  int cur = 0;
  for (int k0 = 0; k0 < Kdim; k0 += 32) {
    const int nxt = cur ^ 1;
    const bool more = (k0 + 32 < Kdim);
    uint4 a0, a1, b0, b1;
    if (more) {                                   // global loads issued early
      a0 = *(const uint4*)(aP + k0 + 32);  a1 = *(const uint4*)(aP + k0 + 40);
      b0 = *(const uint4*)(bP + k0 + 32);  b1 = *(const uint4*)(bP + k0 + 40);
    }
    compute(cur);
    if (more) {
      *(uint4*)&As[nxt][r][c] = a0;  *(uint4*)&As[nxt][r][c + 8] = a1;
      *(uint4*)&Bs[nxt][r][c] = b0;  *(uint4*)&Bs[nxt][r][c + 8] = b1;
    }
    __syncthreads();
    cur = nxt;
  }
#endif

  // C/D layout: VGPR v -> M = v + 8*(lane/16), N = lane%16
  const int mh = (lane >> 4) << 3;
  const int nl = lane & 15;
#pragma unroll
  for (int i = 0; i < 2; ++i)
#pragma unroll
    for (int j = 0; j < 4; ++j)
#pragma unroll
      for (int v = 0; v < 8; ++v)
        epi(wm + 16 * i + mh + v, wn + 16 * j + nl, acc[i][j][v]);
}

// ---------------------------------------------------------------------------
// Elementwise / transpose / reduction kernels
// ---------------------------------------------------------------------------
__global__ __launch_bounds__(256) void k_cast(const float* __restrict__ in,
                                              bf16* __restrict__ out, long n) {
  const long i = (long)blockIdx.x * 256 + threadIdx.x;
  if (i < n) out[i] = (bf16)in[i];
}

// Cast + transpose: in [R,C] fp32 (per blockIdx.z matrix) -> out [C,R] bf16
__global__ __launch_bounds__(256) void k_castT(const float* __restrict__ in,
                                               bf16* __restrict__ out, int R, int C) {
  __shared__ float tile[32][33];
  const float* src = in  + (long)blockIdx.z * R * C;
  bf16*        dst = out + (long)blockIdx.z * R * C;
  const int r0 = blockIdx.x * 32, c0 = blockIdx.y * 32;
  const int tx = threadIdx.x & 31, ty = threadIdx.x >> 5;
#pragma unroll
  for (int i = 0; i < 32; i += 8)
    tile[ty + i][tx] = src[(long)(r0 + ty + i) * C + c0 + tx];
  __syncthreads();
#pragma unroll
  for (int i = 0; i < 32; i += 8)
    dst[(long)(c0 + ty + i) * R + r0 + tx] = (bf16)tile[tx][ty + i];
}

// bf16 transpose per matrix: in [S,E] -> out [E,S]  (V -> V^T)
__global__ __launch_bounds__(256) void k_transB(const bf16* __restrict__ in,
                                                bf16* __restrict__ out) {
  __shared__ bf16 tile[32][34];
  const bf16* src = in  + (long)blockIdx.z * S_ * E_;
  bf16*       dst = out + (long)blockIdx.z * S_ * E_;
  const int s0 = blockIdx.x * 32, f0 = blockIdx.y * 32;
  const int tx = threadIdx.x & 31, ty = threadIdx.x >> 5;
#pragma unroll
  for (int i = 0; i < 32; i += 8)
    tile[ty + i][tx] = src[(long)(s0 + ty + i) * E_ + f0 + tx];
  __syncthreads();
#pragma unroll
  for (int i = 0; i < 32; i += 8)
    dst[(long)(f0 + ty + i) * S_ + s0 + tx] = tile[tx][ty + i];
}

// In-place LayerNorm over last dim (512), per-head gamma/beta, bf16 data
__global__ __launch_bounds__(256) void k_ln_head(bf16* __restrict__ data,
                                                 const float* __restrict__ g,
                                                 const float* __restrict__ bt) {
  const long row = blockIdx.x;                 // B*H*S rows
  const int  h   = (int)((row / S_) % H_);
  bf16* p = data + row * (long)D_;
  const int t = threadIdx.x;
  float x0 = (float)p[t], x1 = (float)p[t + 256];
  __shared__ float sh[256];
  sh[t] = x0 + x1;
  __syncthreads();
  for (int o = 128; o > 0; o >>= 1) { if (t < o) sh[t] += sh[t + o]; __syncthreads(); }
  const float mu = sh[0] * (1.0f / 512.0f);
  __syncthreads();
  const float d0 = x0 - mu, d1 = x1 - mu;
  sh[t] = d0 * d0 + d1 * d1;
  __syncthreads();
  for (int o = 128; o > 0; o >>= 1) { if (t < o) sh[t] += sh[t + o]; __syncthreads(); }
  const float rs = rsqrtf(sh[0] * (1.0f / 512.0f) + kEps);
  p[t]       = (bf16)(d0 * rs * g[h * D_ + t]       + bt[h * D_ + t]);
  p[t + 256] = (bf16)(d1 * rs * g[h * D_ + t + 256] + bt[h * D_ + t + 256]);
}

// In-place softmax over rows of length S (1024), bf16
__global__ __launch_bounds__(256) void k_softmax(bf16* __restrict__ P) {
  const long row = blockIdx.x;
  bf16* p = P + row * (long)S_;
  const int t = threadIdx.x;
  float f[4];
#pragma unroll
  for (int i = 0; i < 4; ++i) f[i] = (float)p[t + 256 * i];
  __shared__ float sh[256];
  sh[t] = fmaxf(fmaxf(f[0], f[1]), fmaxf(f[2], f[3]));
  __syncthreads();
  for (int o = 128; o > 0; o >>= 1) { if (t < o) sh[t] = fmaxf(sh[t], sh[t + o]); __syncthreads(); }
  const float rm = sh[0];
  __syncthreads();
  float s = 0.f;
#pragma unroll
  for (int i = 0; i < 4; ++i) { f[i] = __expf(f[i] - rm); s += f[i]; }
  sh[t] = s;
  __syncthreads();
  for (int o = 128; o > 0; o >>= 1) { if (t < o) sh[t] += sh[t + o]; __syncthreads(); }
  const float inv = 1.0f / sh[0];
#pragma unroll
  for (int i = 0; i < 4; ++i) p[t + 256 * i] = (bf16)(f[i] * inv);
}

// out = LN(x + mlp), fp32
__global__ __launch_bounds__(256) void k_final(const float* __restrict__ x,
                                               const float* __restrict__ mlp,
                                               const float* __restrict__ g,
                                               const float* __restrict__ bt,
                                               float* __restrict__ out) {
  const long row = blockIdx.x;                 // B*S rows
  const int  t   = threadIdx.x;
  const float* px = x   + row * (long)E_;
  const float* pm = mlp + row * (long)E_;
  float x0 = px[t] + pm[t];
  float x1 = px[t + 256] + pm[t + 256];
  __shared__ float sh[256];
  sh[t] = x0 + x1;
  __syncthreads();
  for (int o = 128; o > 0; o >>= 1) { if (t < o) sh[t] += sh[t + o]; __syncthreads(); }
  const float mu = sh[0] * (1.0f / 512.0f);
  __syncthreads();
  const float d0 = x0 - mu, d1 = x1 - mu;
  sh[t] = d0 * d0 + d1 * d1;
  __syncthreads();
  for (int o = 128; o > 0; o >>= 1) { if (t < o) sh[t] += sh[t + o]; __syncthreads(); }
  const float rs = rsqrtf(sh[0] * (1.0f / 512.0f) + kEps);
  out[row * (long)E_ + t]       = d0 * rs * g[t]       + bt[t];
  out[row * (long)E_ + t + 256] = d1 * rs * g[t + 256] + bt[t + 256];
}

// ---------------------------------------------------------------------------
// GEMM wrapper kernels (all TRANSB: weights/K/V^T pre-transposed)
// ---------------------------------------------------------------------------
// Q/K/V projection: out[b,h,s,:] = x[b,s,:] @ W[h] + bias[h];  Wt is [H,D,E]
__global__ __launch_bounds__(256) void k_proj(const bf16* __restrict__ xb,
                                              const bf16* __restrict__ Wt,
                                              const float* __restrict__ bias,
                                              bf16* __restrict__ out) {
  const int  h  = blockIdx.z;
  const long m0 = (long)blockIdx.x * 128;
  const int  n0 = blockIdx.y * 128;
  const bf16* Ag = xb + m0 * E_;
  const bf16* Bg = Wt + ((long)h * D_ + n0) * E_;
  gemm128x128(Ag, E_, Bg, E_, E_,
    [&](int ml, int nl, float v) {
      const long m = m0 + ml;
      const int  n = n0 + nl;
      const long b = m >> 10;          // / S_
      const long s = m & (S_ - 1);
      out[(((b * H_) + h) * (long)S_ + s) * D_ + n] = (bf16)(v + bias[h * D_ + n]);
    });
}

// scores = Q K^T / 22 -> bf16 P (pre-softmax); K natural layout is already K^T operand
__global__ __launch_bounds__(256) void k_scores(const bf16* __restrict__ Q,
                                                const bf16* __restrict__ Km,
                                                bf16* __restrict__ P) {
  const int  bh = blockIdx.z;
  const long m0 = (long)blockIdx.x * 128;
  const int  n0 = blockIdx.y * 128;
  const bf16* Ag = Q  + ((long)bh * S_ + m0) * D_;
  const bf16* Bg = Km + ((long)bh * S_ + n0) * D_;
  gemm128x128(Ag, D_, Bg, D_, D_,
    [&](int ml, int nl, float v) {
      P[((long)bh * S_ + (m0 + ml)) * S_ + (n0 + nl)] = (bf16)(v * kInvScale);
    });
}

// A = P @ V via V^T [B,H,E,S], scattered into concat layout [B,S,H,E]
__global__ __launch_bounds__(256) void k_pv(const bf16* __restrict__ P,
                                            const bf16* __restrict__ Vt,
                                            bf16* __restrict__ Acat) {
  const int  bh = blockIdx.z;
  const int  b = bh >> 3, h = bh & 7;
  const long m0 = (long)blockIdx.x * 128;
  const int  n0 = blockIdx.y * 128;
  const bf16* Ag = P  + ((long)bh * S_ + m0) * S_;
  const bf16* Bg = Vt + ((long)bh * E_ + n0) * S_;
  gemm128x128(Ag, S_, Bg, S_, S_,
    [&](int ml, int nl, float v) {
      const long s = m0 + ml;
      const int  n = n0 + nl;
      Acat[(((long)b * S_ + s) * H_ + h) * E_ + n] = (bf16)v;
    });
}

// hid = relu(Acat @ W1 + b1);  W1t is [E, H*E]
__global__ __launch_bounds__(256) void k_mlp1(const bf16* __restrict__ Acat,
                                              const bf16* __restrict__ W1t,
                                              const float* __restrict__ b1,
                                              bf16* __restrict__ hid) {
  const long m0 = (long)blockIdx.x * 128;
  const int  n0 = blockIdx.y * 128;
  gemm128x128(Acat + m0 * (H_ * E_), H_ * E_, W1t + (long)n0 * (H_ * E_), H_ * E_, H_ * E_,
    [&](int ml, int nl, float v) {
      const int n = n0 + nl;
      hid[(m0 + ml) * (long)E_ + n] = (bf16)fmaxf(v + b1[n], 0.f);
    });
}

// mlp = relu(hid @ W2 + b2), fp32;  W2t is [E, E]
__global__ __launch_bounds__(256) void k_mlp2(const bf16* __restrict__ hid,
                                              const bf16* __restrict__ W2t,
                                              const float* __restrict__ b2,
                                              float* __restrict__ mlp) {
  const long m0 = (long)blockIdx.x * 128;
  const int  n0 = blockIdx.y * 128;
  gemm128x128(hid + m0 * E_, E_, W2t + (long)n0 * E_, E_, E_,
    [&](int ml, int nl, float v) {
      const int n = n0 + nl;
      mlp[(m0 + ml) * (long)E_ + n] = fmaxf(v + b2[n], 0.f);
    });
}

// ---------------------------------------------------------------------------
extern "C" void kernel_launch(void* const* d_in, const int* in_sizes, int n_in,
                              void* d_out, int out_size, void* d_ws, size_t ws_size,
                              hipStream_t stream) {
  (void)in_sizes; (void)n_in; (void)out_size; (void)ws_size;
  const float* x   = (const float*)d_in[0];
  const float* Wq  = (const float*)d_in[1];
  const float* bq  = (const float*)d_in[2];
  const float* Wk  = (const float*)d_in[3];
  const float* bk  = (const float*)d_in[4];
  const float* Wv  = (const float*)d_in[5];
  const float* bv  = (const float*)d_in[6];
  const float* qg  = (const float*)d_in[7];
  const float* qb  = (const float*)d_in[8];
  const float* kg  = (const float*)d_in[9];
  const float* kb  = (const float*)d_in[10];
  const float* vg  = (const float*)d_in[11];
  const float* vb  = (const float*)d_in[12];
  const float* W1  = (const float*)d_in[13];
  const float* b1  = (const float*)d_in[14];
  const float* W2  = (const float*)d_in[15];
  const float* b2  = (const float*)d_in[16];
  const float* lng = (const float*)d_in[17];
  const float* lnb = (const float*)d_in[18];
  float* out = (float*)d_out;

  char*  ws  = (char*)d_ws;
  size_t off = 0;
  auto alloc = [&](size_t bytes) -> char* {
    char* p = ws + off;
    off += (bytes + 255) & ~(size_t)255;
    return p;
  };
  bf16*  xb  = (bf16*)alloc((size_t)B_ * S_ * E_ * 2);
  bf16*  Wqt = (bf16*)alloc((size_t)H_ * E_ * D_ * 2);   // [H,D,E]
  bf16*  Wkt = (bf16*)alloc((size_t)H_ * E_ * D_ * 2);
  bf16*  Wvt = (bf16*)alloc((size_t)H_ * E_ * E_ * 2);
  bf16*  W1t = (bf16*)alloc((size_t)H_ * E_ * E_ * 2);   // [E, H*E]
  bf16*  W2t = (bf16*)alloc((size_t)E_ * E_ * 2);        // [E, E]
  bf16*  Qb  = (bf16*)alloc((size_t)B_ * H_ * S_ * D_ * 2);
  bf16*  Kb  = (bf16*)alloc((size_t)B_ * H_ * S_ * D_ * 2);
  bf16*  Vb  = (bf16*)alloc((size_t)B_ * H_ * S_ * E_ * 2);
  bf16*  Vt  = (bf16*)alloc((size_t)B_ * H_ * E_ * S_ * 2);
  bf16*  P   = (bf16*)alloc((size_t)B_ * H_ * S_ * S_ * 2);
  bf16*  Ac  = (bf16*)alloc((size_t)B_ * S_ * H_ * E_ * 2);
  bf16*  hid = (bf16*)alloc((size_t)B_ * S_ * E_ * 2);
  float* mlp = (float*)alloc((size_t)B_ * S_ * E_ * 4);

  const dim3 blk(256);
  {
    const long n = (long)B_ * S_ * E_;
    k_cast<<<dim3((unsigned)((n + 255) / 256)), blk, 0, stream>>>(x, xb, n);
  }
  k_castT<<<dim3(E_ / 32, D_ / 32, H_), blk, 0, stream>>>(Wq, Wqt, E_, D_);
  k_castT<<<dim3(E_ / 32, D_ / 32, H_), blk, 0, stream>>>(Wk, Wkt, E_, D_);
  k_castT<<<dim3(E_ / 32, E_ / 32, H_), blk, 0, stream>>>(Wv, Wvt, E_, E_);
  k_castT<<<dim3(H_ * E_ / 32, E_ / 32, 1), blk, 0, stream>>>(W1, W1t, H_ * E_, E_);
  k_castT<<<dim3(E_ / 32, E_ / 32, 1), blk, 0, stream>>>(W2, W2t, E_, E_);

  const dim3 gproj(B_ * S_ / 128, D_ / 128, H_);   // 128 x 4 x 8
  k_proj<<<gproj, blk, 0, stream>>>(xb, Wqt, bq, Qb);
  k_proj<<<gproj, blk, 0, stream>>>(xb, Wkt, bk, Kb);
  k_proj<<<gproj, blk, 0, stream>>>(xb, Wvt, bv, Vb);

  const dim3 gln(B_ * H_ * S_);
  k_ln_head<<<gln, blk, 0, stream>>>(Qb, qg, qb);
  k_ln_head<<<gln, blk, 0, stream>>>(Kb, kg, kb);
  k_ln_head<<<gln, blk, 0, stream>>>(Vb, vg, vb);

  k_transB<<<dim3(S_ / 32, E_ / 32, B_ * H_), blk, 0, stream>>>(Vb, Vt);

  const dim3 gsc(S_ / 128, S_ / 128, B_ * H_);     // 8 x 8 x 128
  k_scores<<<gsc, blk, 0, stream>>>(Qb, Kb, P);

  k_softmax<<<dim3(B_ * H_ * S_), blk, 0, stream>>>(P);

  const dim3 gpv(S_ / 128, E_ / 128, B_ * H_);     // 8 x 4 x 128
  k_pv<<<gpv, blk, 0, stream>>>(P, Vt, Ac);

  const dim3 gm(B_ * S_ / 128, E_ / 128, 1);       // 128 x 4
  k_mlp1<<<gm, blk, 0, stream>>>(Ac, W1t, b1, hid);
  k_mlp2<<<gm, blk, 0, stream>>>(hid, W2t, b2, mlp);

  k_final<<<dim3(B_ * S_), blk, 0, stream>>>(x, mlp, lng, lnb, out);
}
